// MambaBlock_8667244003586
// MI455X (gfx1250) — matrix operand order repaired and checked by reading, not compiled, not run on hardware
//
#include <hip/hip_runtime.h>
#include <hip/hip_bf16.h>
#include <math.h>

typedef __attribute__((ext_vector_type(16))) _Float16 v16h;
typedef __attribute__((ext_vector_type(8)))  _Float16 v8h;
typedef __attribute__((ext_vector_type(8)))  float    v8f;
typedef int v4i __attribute__((vector_size(16)));

#define BB   2
#define LL   4096
#define DM   768
#define EE   1536
#define MM   (BB*LL)      // 8192 rows
#define NS   16
#define NXP  800          // 784 (W_xp) + 16 (W_dt)
#define NCH  64
#define CHL  64           // NCH*CHL == LL

#if __has_builtin(__builtin_amdgcn_global_load_async_to_lds_b128)
#define HAVE_ASYNC_LDS 1
#else
#define HAVE_ASYNC_LDS 0
#endif

#if HAVE_ASYNC_LDS
__device__ __forceinline__ void gld_async_b128(const _Float16* g, _Float16* l) {
  __builtin_amdgcn_global_load_async_to_lds_b128(
      (v4i*)g, (v4i*)l, 0 /*imm offset*/, 0 /*cpol*/);
}
__device__ __forceinline__ void wait_async0() {
#if __has_builtin(__builtin_amdgcn_s_wait_asynccnt)
  __builtin_amdgcn_s_wait_asynccnt(0);
#else
  asm volatile("s_wait_asynccnt 0" ::: "memory");
#endif
}
#else
__device__ __forceinline__ void gld_async_b128(const _Float16* g, _Float16* l) {
  *(uint4*)l = *(const uint4*)g;     // sync fallback: load->VGPR->LDS
}
__device__ __forceinline__ void wait_async0() {}
#endif

// ---------------------------------------------------------------------------
// f16 WMMA GEMM: C[M,N] = A[M,K] * Bt[N,K]^T + bias[N]  (f32 out)
// 256 threads (8 waves), block tile 128x128, wave tile 32x64, K-step 32.
// Double-buffered LDS staged with GLOBAL_LOAD_ASYNC_TO_LDS_B128 (ASYNCcnt).
// ---------------------------------------------------------------------------
template <int K, bool NGUARD>
__global__ __launch_bounds__(256) void gemm_f16_wmma(
    const _Float16* __restrict__ A,
    const _Float16* __restrict__ Bt,
    const float*    __restrict__ bias,
    float*          __restrict__ C,
    int M, int N)
{
  constexpr int NT  = K / 32;      // K-steps
  constexpr int LDW = 40;          // LDS row stride (f16) -> 80B, 16B aligned

  __shared__ __align__(16) _Float16 sA[2][128*LDW];
  __shared__ __align__(16) _Float16 sB[2][128*LDW];

  const int tid   = threadIdx.x;
  const int lane  = tid & 31;
  const int wid   = tid >> 5;
  const int wm    = wid & 3;       // 4 waves along M
  const int wn    = wid >> 2;      // 2 waves along N
  const int lmod  = lane & 15;
  const int lhalf = lane >> 4;
  const int m0 = blockIdx.x * 128;
  const int n0 = blockIdx.y * 128;

  // Per-thread staging addresses: chunk0 = (row, c8), chunk1 = (row+64, c8)
  const int srow = tid >> 2;
  const int sc8  = (tid & 3) << 3;
  const _Float16* gA = A  + (size_t)(m0 + srow)*K + sc8;
  const _Float16* gB = Bt + (size_t)(n0 + srow)*K + sc8;
  const int lofs  = srow*LDW + sc8;
  const bool bg0  = !NGUARD || (n0 + srow)      < N;
  const bool bg1  = !NGUARD || (n0 + srow + 64) < N;

  v8f acc[2][4] = {};

  auto stage = [&](int buf, int kk) {
    gld_async_b128(gA + kk,                &sA[buf][lofs]);
    gld_async_b128(gA + (size_t)64*K + kk, &sA[buf][lofs + 64*LDW]);
    if (bg0) gld_async_b128(gB + kk,                &sB[buf][lofs]);
    if (bg1) gld_async_b128(gB + (size_t)64*K + kk, &sB[buf][lofs + 64*LDW]);
  };

  stage(0, 0);

#pragma unroll 1
  for (int t = 0; t < NT; ++t) {
    const int cur = t & 1;
    wait_async0();            // tile t resident in LDS (this wave's asyncs)
    __syncthreads();          // ...and visible to all waves; prior reads done
    if (t + 1 < NT) stage(cur ^ 1, (t + 1) * 32);

    // A fragments: lane holds row M=lmod (+16/m-tile); K halves per lhalf
    v16h af[2];
#pragma unroll
    for (int mt = 0; mt < 2; ++mt) {
      const _Float16* p = &sA[cur][(wm*32 + mt*16 + lmod)*LDW + lhalf*8];
      v8h lo = *(const v8h*)p;          // K = lhalf*8 .. +7
      v8h hi = *(const v8h*)(p + 16);   // K = 16 + lhalf*8 .. +7
      af[mt] = __builtin_shufflevector(lo, hi, 0,1,2,3,4,5,6,7,8,9,10,11,12,13,14,15);
    }
    // B fragments: lane holds col N=lmod; K halves per lhalf (same as A)
    v16h bf[4];
#pragma unroll
    for (int nt = 0; nt < 4; ++nt) {
      const _Float16* p = &sB[cur][(wn*64 + nt*16 + lmod)*LDW + lhalf*8];
      v8h lo = *(const v8h*)p;
      v8h hi = *(const v8h*)(p + 16);
      bf[nt] = __builtin_shufflevector(lo, hi, 0,1,2,3,4,5,6,7,8,9,10,11,12,13,14,15);
    }
#pragma unroll
    for (int mt = 0; mt < 2; ++mt)
#pragma unroll
      for (int nt = 0; nt < 4; ++nt)
        acc[mt][nt] = __builtin_amdgcn_wmma_f32_16x16x32_f16(
            false, af[mt], false, bf[nt], (short)0, acc[mt][nt], false, false);

    __syncthreads();          // all reads of buf `cur` done before t+2 stages it
  }

  // C/D layout: c[r] -> row r + 8*lhalf, col lmod
#pragma unroll
  for (int mt = 0; mt < 2; ++mt)
#pragma unroll
    for (int nt = 0; nt < 4; ++nt) {
      int gn = n0 + wn*64 + nt*16 + lmod;
      if (!NGUARD || gn < N) {
        int rowb = m0 + wm*32 + mt*16 + lhalf*8;
        float bv = bias[gn];
#pragma unroll
        for (int r = 0; r < 8; ++r)
          C[(size_t)(rowb + r)*N + gn] = acc[mt][nt][r] + bv;
      }
    }
}

// ---------------------------------------------------------------------------
// Transpose + f32->f16: src[K,N] -> dst[N,K]
// ---------------------------------------------------------------------------
__global__ void trcvt_kernel(const float* __restrict__ src,
                             _Float16* __restrict__ dst, int K, int N)
{
  int idx = blockIdx.x*256 + threadIdx.x;
  if (idx >= K*N) return;
  int k = idx / N, n = idx - k*N;
  dst[(size_t)n*K + k] = (_Float16)src[idx];
}

__global__ void bcat_kernel(const float* __restrict__ bxp,
                            const float* __restrict__ bdt,
                            float* __restrict__ out)
{
  int i = blockIdx.x*256 + threadIdx.x;
  if (i < NXP) out[i] = (i < 784) ? bxp[i] : bdt[i - 784];
}

// ---------------------------------------------------------------------------
// LayerNorm over 768, f16 output. 256 threads x 3 elems.
// ---------------------------------------------------------------------------
__global__ __launch_bounds__(256) void ln_kernel(
    const float* __restrict__ x, const float* __restrict__ g,
    const float* __restrict__ b, _Float16* __restrict__ xn)
{
  __shared__ float rs[256], rq[256];
  int row = blockIdx.x, tid = threadIdx.x;
  const float* xr = x + (size_t)row*DM;
  float v0 = xr[tid], v1 = xr[tid+256], v2 = xr[tid+512];
  rs[tid] = v0 + v1 + v2;
  rq[tid] = v0*v0 + v1*v1 + v2*v2;
  __syncthreads();
  for (int off = 128; off > 0; off >>= 1) {
    if (tid < off) { rs[tid] += rs[tid+off]; rq[tid] += rq[tid+off]; }
    __syncthreads();
  }
  float mu   = rs[0] * (1.f/DM);
  float var  = rq[0] * (1.f/DM) - mu*mu;
  float rstd = rsqrtf(var + 1e-5f);
  _Float16* o = xn + (size_t)row*DM;
  o[tid]     = (_Float16)((v0-mu)*rstd*g[tid]     + b[tid]);
  o[tid+256] = (_Float16)((v1-mu)*rstd*g[tid+256] + b[tid+256]);
  o[tid+512] = (_Float16)((v2-mu)*rstd*g[tid+512] + b[tid+512]);
}

// ---------------------------------------------------------------------------
// Depthwise causal conv (k=4) + SiLU -> f16
// ---------------------------------------------------------------------------
__global__ __launch_bounds__(256) void conv_silu_kernel(
    const float* __restrict__ xin, const float* __restrict__ Wc,
    const float* __restrict__ bc, _Float16* __restrict__ xact)
{
  int e   = blockIdx.y*256 + threadIdx.x;
  int row = blockIdx.x;            // b*L + l
  int l   = row & (LL - 1);
  float acc = bc[e];
#pragma unroll
  for (int j = 0; j < 4; ++j) {
    int lp = l - 3 + j;
    if (lp >= 0)
      acc += Wc[e*4 + j] * xin[(size_t)(row - 3 + j)*EE + e];
  }
  float sv = acc / (1.f + expf(-acc));   // silu
  xact[(size_t)row*EE + e] = (_Float16)sv;
}

// ---------------------------------------------------------------------------
// Elementwise state prep
// ---------------------------------------------------------------------------
__global__ void ew_kernel(const float* __restrict__ xd,
                          float* __restrict__ aA, float* __restrict__ bA,
                          float* __restrict__ dtA, float* __restrict__ dnA)
{
  int idx = blockIdx.x*256 + threadIdx.x;    // MM*NS total
  int row = idx >> 4, s = idx & 15;
  float ds  = xd[(size_t)row*NXP + DM + s];
  float dtl = xd[(size_t)row*NXP + DM + 16 + s];
  float dt  = (dtl > 20.f) ? dtl : log1pf(expf(dtl));
  aA[idx]  = expf(-dt);
  bA[idx]  = ds * expf(-0.5f*dt);
  dtA[idx] = dt;
  float dn = expm1f(-dt);
  dnA[idx] = (dn > -1e-30f) ? -1e-30f : dn;
}

// ---------------------------------------------------------------------------
// Chunked linear recurrence v_t = a_t*v_{t-1} + b_t over L, 32 series.
// ---------------------------------------------------------------------------
__global__ void scan1_kernel(const float* __restrict__ aA,
                             const float* __restrict__ bA,
                             float* __restrict__ vA,
                             float* __restrict__ cAp, float* __restrict__ cVp)
{
  int c = blockIdx.x, lane = threadIdx.x;
  int b = lane >> 4, s = lane & 15;
  float v = 0.f, p = 1.f;
  for (int t = 0; t < CHL; ++t) {
    size_t idx = (size_t)(b*LL + c*CHL + t)*NS + s;
    float a = aA[idx];
    v = a*v + bA[idx];
    p *= a;
    vA[idx] = v;
  }
  cAp[c*32 + lane] = p;
  cVp[c*32 + lane] = v;
}

__global__ void scan2_kernel(const float* __restrict__ cAp,
                             const float* __restrict__ cVp,
                             float* __restrict__ cIn)
{
  int lane = threadIdx.x;
  float carry = 0.f;
  for (int c = 0; c < NCH; ++c) {
    cIn[c*32 + lane] = carry;
    carry = cAp[c*32 + lane]*carry + cVp[c*32 + lane];
  }
}

__global__ void scan3_kernel(const float* __restrict__ aA,
                             float* __restrict__ vA,
                             const float* __restrict__ cIn)
{
  int c = blockIdx.x, lane = threadIdx.x;
  int b = lane >> 4, s = lane & 15;
  float cin = cIn[c*32 + lane];
  float p = 1.f;
  for (int t = 0; t < CHL; ++t) {
    size_t idx = (size_t)(b*LL + c*CHL + t)*NS + s;
    p *= aA[idx];
    vA[idx] += p*cin;
  }
}

// ---------------------------------------------------------------------------
// Build z[M,1536] f16: low half = xs + us@W_us + b_us, high half = x_act hi
// ---------------------------------------------------------------------------
__global__ __launch_bounds__(256) void yz_kernel(
    const float* __restrict__ xd, const float* __restrict__ dtA,
    const float* __restrict__ bA, const float* __restrict__ dnA,
    const float* __restrict__ vA,
    const float* __restrict__ Wus, const float* __restrict__ bus,
    const _Float16* __restrict__ xact, _Float16* __restrict__ z)
{
  __shared__ float sus[NS];
  int row = blockIdx.x, tid = threadIdx.x;
  int l = row & (LL - 1);
  if (tid < NS) {
    size_t idx = (size_t)row*NS + tid;
    float dt = dtA[idx];
    float kf = (float)(LL - 1 - l);
    float e1 = expf(-kf*dt);
    float em = expm1f(-kf*dt);
    sus[tid] = e1*vA[idx] + bA[idx]*em/dnA[idx];
  }
  __syncthreads();
  for (int j = tid; j < DM; j += 256) {
    float y = xd[(size_t)row*NXP + j] + bus[j];
#pragma unroll
    for (int s = 0; s < NS; ++s) y += sus[s] * Wus[s*DM + j];
    z[(size_t)row*EE + j]      = (_Float16)y;
    z[(size_t)row*EE + DM + j] = xact[(size_t)row*EE + DM + j];
  }
}

// ---------------------------------------------------------------------------
extern "C" void kernel_launch(void* const* d_in, const int* in_sizes, int n_in,
                              void* d_out, int out_size, void* d_ws, size_t ws_size,
                              hipStream_t stream)
{
  const float* x      = (const float*)d_in[0];
  const float* ln_g   = (const float*)d_in[1];
  const float* ln_b   = (const float*)d_in[2];
  const float* W_in   = (const float*)d_in[3];
  const float* b_in   = (const float*)d_in[4];
  const float* W_conv = (const float*)d_in[5];
  const float* b_conv = (const float*)d_in[6];
  const float* W_xp   = (const float*)d_in[7];
  const float* b_xp   = (const float*)d_in[8];
  const float* W_dt   = (const float*)d_in[9];
  const float* b_dt   = (const float*)d_in[10];
  const float* W_us   = (const float*)d_in[11];
  const float* b_us   = (const float*)d_in[12];
  const float* W_out  = (const float*)d_in[13];
  const float* b_out  = (const float*)d_in[14];
  float* out = (float*)d_out;

  char* w = (char*)d_ws;
  size_t off = 0;
  auto alloc = [&](size_t bytes) -> char* {
    char* p = w + off;
    off = (off + bytes + 255) & ~(size_t)255;
    return p;
  };

  _Float16* xn    = (_Float16*)alloc((size_t)MM*DM*2);
  _Float16* WinT  = (_Float16*)alloc((size_t)EE*DM*2);
  _Float16* WcatT = (_Float16*)alloc((size_t)NXP*EE*2);
  _Float16* WoutT = (_Float16*)alloc((size_t)DM*EE*2);
  float*    bcat  = (float*)alloc((size_t)NXP*4);
  float*    x_in  = (float*)alloc((size_t)MM*EE*4);   // conv input; reused as xd
  _Float16* xact  = (_Float16*)alloc((size_t)MM*EE*2);
  float*    aArr  = (float*)alloc((size_t)MM*NS*4);
  float*    bArr  = (float*)alloc((size_t)MM*NS*4);
  float*    dtArr = (float*)alloc((size_t)MM*NS*4);
  float*    dnArr = (float*)alloc((size_t)MM*NS*4);
  float*    vArr  = (float*)alloc((size_t)MM*NS*4);
  float*    cA    = (float*)alloc((size_t)NCH*32*4);
  float*    cV    = (float*)alloc((size_t)NCH*32*4);
  float*    cIn   = (float*)alloc((size_t)NCH*32*4);
  _Float16* z     = (_Float16*)alloc((size_t)MM*EE*2);
  float*    xd    = x_in;   // alias: x_in fully consumed by conv before GEMM2

  // Weight prep (f32 -> f16, transposed to N x K)
  trcvt_kernel<<<(DM*EE + 255)/256, 256, 0, stream>>>(W_in,  WinT,  DM, EE);
  trcvt_kernel<<<(EE*784 + 255)/256, 256, 0, stream>>>(W_xp,  WcatT, EE, 784);
  trcvt_kernel<<<(EE*16  + 255)/256, 256, 0, stream>>>(W_dt,  WcatT + (size_t)784*EE, EE, 16);
  trcvt_kernel<<<(EE*DM  + 255)/256, 256, 0, stream>>>(W_out, WoutT, EE, DM);
  bcat_kernel<<<4, 256, 0, stream>>>(b_xp, b_dt, bcat);

  // 1) LayerNorm -> f16
  ln_kernel<<<MM, 256, 0, stream>>>(x, ln_g, ln_b, xn);

  // 2) GEMM1: xn @ W_in + b_in -> x_in (f32), K=768
  gemm_f16_wmma<DM, false><<<dim3(MM/128, EE/128), 256, 0, stream>>>(
      xn, WinT, b_in, x_in, MM, EE);

  // 3) depthwise conv + SiLU -> f16
  conv_silu_kernel<<<dim3(MM, EE/256), 256, 0, stream>>>(x_in, W_conv, b_conv, xact);

  // 4) GEMM2: x_act @ [W_xp | W_dt] + [b_xp | b_dt] -> xd (f32, 800 cols), K=1536
  gemm_f16_wmma<EE, true><<<dim3(MM/128, (NXP + 127)/128), 256, 0, stream>>>(
      xact, WcatT, bcat, xd, MM, NXP);

  // 5) state prep
  ew_kernel<<<(MM*NS)/256, 256, 0, stream>>>(xd, aArr, bArr, dtArr, dnArr);

  // 6) chunked scan (32 series x 4096)
  scan1_kernel<<<NCH, 32, 0, stream>>>(aArr, bArr, vArr, cA, cV);
  scan2_kernel<<<1, 32, 0, stream>>>(cA, cV, cIn);
  scan3_kernel<<<NCH, 32, 0, stream>>>(aArr, vArr, cIn);

  // 7) us, y = xs + us@W_us + b_us; build z = [y | x_act_hi] (f16)
  yz_kernel<<<MM, 256, 0, stream>>>(xd, dtArr, bArr, dnArr, vArr, W_us, b_us, xact, z);

  // 8) GEMM3: z @ W_out + b_out -> output (f32), K=1536
  gemm_f16_wmma<EE, false><<<dim3(MM/128, DM/128), 256, 0, stream>>>(
      z, WoutT, b_out, out, MM, DM);
}
